// YoloxLoss_18210661335461
// MI455X (gfx1250) — matrix operand order repaired and checked by reading, not compiled, not run on hardware
//
#include <hip/hip_runtime.h>
#include <hip/hip_bf16.h>
#include <math.h>

#define NUM_B 32
#define NUM_G 20
#define NUM_C 80
#define C_PAD 96
#define NUM_A 8400
#define A0    6400
#define A1    8000
#define LOSS_BLOCKS ((NUM_A + 255) / 256)

typedef __attribute__((ext_vector_type(16))) _Float16 v16h;
typedef __attribute__((ext_vector_type(8)))  _Float16 v8h;
typedef __attribute__((ext_vector_type(8)))  float    v8f;

__device__ __forceinline__ float sigm(float x) { return 1.0f / (1.0f + expf(-x)); }
__device__ __forceinline__ float bce(float x, float t) {
    return fmaxf(x, 0.0f) - x * t + log1pf(expf(-fabsf(x)));
}

// ---------------------------------------------------------------------------
// K1: per-anchor decode + per-class log-prob precompute + per-GT IoU / masks.
// Writes: bbox[a], slp[a]=sum_c l1p, Dh[a][96]=(lp-l1p) in f16 (zero padded),
//         iou[g*A+a], inb[a]=in_box&in_ctr bitmask, fgc[a]=fg_cand flag.
// ---------------------------------------------------------------------------
__global__ void __launch_bounds__(256)
k_decode(const float* __restrict__ p8, const float* __restrict__ p16,
         const float* __restrict__ p32, const float* __restrict__ labels, int b,
         float4* __restrict__ bbox, float* __restrict__ slp,
         _Float16* __restrict__ Dh, float* __restrict__ iou,
         unsigned* __restrict__ inb, unsigned* __restrict__ fgc)
{
    int a = blockIdx.x * blockDim.x + threadIdx.x;
    if (a >= NUM_A) return;
    const float* base; int i, W, HW; float s;
    if (a < A0)      { base = p8  + (size_t)b * 85 * 6400; i = a;      W = 80; HW = 6400; s = 8.f;  }
    else if (a < A1) { base = p16 + (size_t)b * 85 * 1600; i = a - A0; W = 40; HW = 1600; s = 16.f; }
    else             { base = p32 + (size_t)b * 85 * 400;  i = a - A1; W = 20; HW = 400;  s = 32.f; }
    int w = i % W, h = i / W;

    float cx = (base[0 * HW + i] + (float)w) * s;
    float cy = (base[1 * HW + i] + (float)h) * s;
    float bw = expf(base[2 * HW + i]) * s;
    float bh = expf(base[3 * HW + i]) * s;
    bbox[a] = make_float4(cx, cy, bw, bh);

    float so = sigm(base[4 * HW + i]);
    float acc_l1p = 0.f;
    for (int c = 0; c < NUM_C; ++c) {
        float x   = base[(5 + c) * HW + i];
        float p   = sqrtf(sigm(x) * so);
        float lp  = fmaxf(logf(p),    -100.f);
        float l1p = fmaxf(log1pf(-p), -100.f);
        acc_l1p  += l1p;
        Dh[(size_t)a * C_PAD + c] = (_Float16)(lp - l1p);
    }
    for (int c = NUM_C; c < C_PAD; ++c) Dh[(size_t)a * C_PAD + c] = (_Float16)0.f;
    slp[a] = acc_l1p;

    float xc = ((float)w + 0.5f) * s;
    float yc = ((float)h + 0.5f) * s;
    unsigned mboth = 0; bool any = false;
    for (int g = 0; g < NUM_G; ++g) {
        const float* L = labels + ((size_t)b * NUM_G + g) * 5;
        float gx = L[0], gy = L[1], gw = L[2], gh = L[3];
        bool ib = (xc > gx - 0.5f * gw) && (xc < gx + 0.5f * gw) &&
                  (yc > gy - 0.5f * gh) && (yc < gy + 0.5f * gh);
        float r = 2.5f * s;
        bool ic = (xc > gx - r) && (xc < gx + r) && (yc > gy - r) && (yc < gy + r);
        if (ib && ic) mboth |= 1u << g;
        any = any || ib || ic;
        float tlx = fmaxf(gx - 0.5f * gw, cx - 0.5f * bw);
        float tly = fmaxf(gy - 0.5f * gh, cy - 0.5f * bh);
        float brx = fminf(gx + 0.5f * gw, cx + 0.5f * bw);
        float bry = fminf(gy + 0.5f * gh, cy + 0.5f * bh);
        float inter = (tlx < brx && tly < bry) ? (brx - tlx) * (bry - tly) : 0.f;
        float u = gw * gh + bw * bh - inter;
        iou[(size_t)g * NUM_A + a] = inter / (u + 1e-16f);
    }
    inb[a] = mboth;
    fgc[a] = any ? 1u : 0u;
}

// ---------------------------------------------------------------------------
// K1b: one-hot class matrix T (32 x 96, f16, rows >= NUM_G zero). One wave;
// built once per batch so the 525 WMMA blocks just load A-fragments from L2.
// ---------------------------------------------------------------------------
__global__ void __launch_bounds__(32)
k_onehot(const float* __restrict__ labels, int b, _Float16* __restrict__ Th)
{
    int g = threadIdx.x;                          // 0..31
    int clsg = (g < NUM_G) ? (int)labels[((size_t)b * NUM_G + g) * 5 + 4] : -1;
    for (int K = 0; K < C_PAD; ++K)
        Th[g * C_PAD + K] = (K == clsg) ? (_Float16)1.0f : (_Float16)0.0f;
}

// ---------------------------------------------------------------------------
// K2: cls_cost einsum via V_WMMA_F32_16X16X32_F16 + full cost assembly.
// One wave per 16-anchor tile. A-fragment (one-hot T) and B-fragment (Dh)
// are both straight vector loads using the same contiguous k order for A and B:
//   A: lane row M=lane%16, elem e <-> K = kc*32 + e + 16*(lane>=16)
//      -> two contiguous 8-half (16 B) runs per chunk.
//   B: lane col N=lane%16, elem e <-> K = kc*32 + e + 16*(lane>=16)
//      -> one contiguous 16-half (32 B) run per chunk.
// ---------------------------------------------------------------------------
__global__ void __launch_bounds__(32)
k_cost_wmma(const _Float16* __restrict__ Th, const _Float16* __restrict__ Dh,
            const float* __restrict__ slp, const float* __restrict__ iou,
            const unsigned* __restrict__ inb, const unsigned* __restrict__ fgc,
            float* __restrict__ cost)
{
    int lane = threadIdx.x;
    int n    = lane & 15;
    bool hi  = lane >= 16;
    int a    = blockIdx.x * 16 + n;

    const _Float16* drow = Dh + (size_t)a * C_PAD + (hi ? 16 : 0);
    v16h bf0 = *(const v16h*)(drow + 0);
    v16h bf1 = *(const v16h*)(drow + 32);
    v16h bf2 = *(const v16h*)(drow + 64);

    float    sl = slp[a];
    unsigned fa = fgc[a];
    unsigned mb = inb[a];

    for (int tM = 0; tM < 2; ++tM) {
        const _Float16* trow = Th + (tM * 16 + n) * C_PAD + (hi ? 16 : 0);
        v8f acc = {};
        #pragma unroll
        for (int kc = 0; kc < 3; ++kc) {
            union { v16h v; v8h h[2]; } u;
            u.h[0] = *(const v8h*)(trow + kc * 32);
            u.h[1] = *(const v8h*)(trow + kc * 32 + 8);
            v16h bf = (kc == 0) ? bf0 : (kc == 1) ? bf1 : bf2;
            acc = __builtin_amdgcn_wmma_f32_16x16x32_f16(
                false, u.v, false, bf, (short)0, acc, false, false);
        }
        #pragma unroll
        for (int r = 0; r < 8; ++r) {
            int g = tM * 16 + r + (hi ? 8 : 0);
            if (g < NUM_G) {
                float iv = iou[(size_t)g * NUM_A + a];
                float im = fa ? iv : 0.f;
                float cc = -(acc[r] + sl)
                         + 3.f * (-logf(im + 1e-8f))
                         + (((mb >> g) & 1u) ? 0.f : 100000.f)
                         + (fa ? 0.f : 1000000000.f);
                cost[(size_t)g * NUM_A + a] = cc;
            }
        }
    }
}

// ---------------------------------------------------------------------------
// K3: dyn_k[g] = clamp(int(sum top-10 masked IoUs), 1, ..). One wave per GT.
// ---------------------------------------------------------------------------
__global__ void __launch_bounds__(32)
k_dynk(const float* __restrict__ iou, const unsigned* __restrict__ fgc,
       int* __restrict__ dynk)
{
    int g = blockIdx.x, lane = threadIdx.x;
    float t[10];
    #pragma unroll
    for (int j = 0; j < 10; ++j) t[j] = 0.f;            // descending
    for (int a = lane; a < NUM_A; a += 32) {
        float v = fgc[a] ? iou[(size_t)g * NUM_A + a] : 0.f;
        if (v > t[9]) {
            int j = 9;
            while (j > 0 && v > t[j - 1]) { t[j] = t[j - 1]; --j; }
            t[j] = v;
        }
    }
    __shared__ float sm[32 * 10];
    #pragma unroll
    for (int j = 0; j < 10; ++j) sm[lane * 10 + j] = t[j];
    __syncthreads();
    if (lane == 0) {
        float m[10];
        #pragma unroll
        for (int j = 0; j < 10; ++j) m[j] = 0.f;
        for (int k = 0; k < 320; ++k) {
            float v = sm[k];
            if (v > m[9]) {
                int j = 9;
                while (j > 0 && v > m[j - 1]) { m[j] = m[j - 1]; --j; }
                m[j] = v;
            }
        }
        float s = 0.f;
        #pragma unroll
        for (int j = 0; j < 10; ++j) s += m[j];
        int k = (int)s; if (k < 1) k = 1;
        dynk[g] = k;
    }
}

// ---------------------------------------------------------------------------
// K4: per GT, bottom-10 (cost, idx) lexicographic (stable-argsort-equivalent).
// ---------------------------------------------------------------------------
__device__ __forceinline__ bool lex_less(float v1, int i1, float v2, int i2) {
    return v1 < v2 || (v1 == v2 && i1 < i2);
}

__global__ void __launch_bounds__(32)
k_sel(const float* __restrict__ cost, int* __restrict__ midx)
{
    int g = blockIdx.x, lane = threadIdx.x;
    float cv[10]; int ci[10];
    #pragma unroll
    for (int j = 0; j < 10; ++j) { cv[j] = INFINITY; ci[j] = 0x7fffffff; }
    for (int a = lane; a < NUM_A; a += 32) {
        float v = cost[(size_t)g * NUM_A + a];
        if (lex_less(v, a, cv[9], ci[9])) {
            int j = 9;
            while (j > 0 && lex_less(v, a, cv[j - 1], ci[j - 1])) {
                cv[j] = cv[j - 1]; ci[j] = ci[j - 1]; --j;
            }
            cv[j] = v; ci[j] = a;
        }
    }
    __shared__ float scv[320];
    __shared__ int   sci[320];
    #pragma unroll
    for (int j = 0; j < 10; ++j) { scv[lane * 10 + j] = cv[j]; sci[lane * 10 + j] = ci[j]; }
    __syncthreads();
    if (lane == 0) {
        float mv[10]; int mi[10];
        #pragma unroll
        for (int j = 0; j < 10; ++j) { mv[j] = INFINITY; mi[j] = 0x7fffffff; }
        for (int k = 0; k < 320; ++k) {
            float v = scv[k]; int id = sci[k];
            if (lex_less(v, id, mv[9], mi[9])) {
                int j = 9;
                while (j > 0 && lex_less(v, id, mv[j - 1], mi[j - 1])) {
                    mv[j] = mv[j - 1]; mi[j] = mi[j - 1]; --j;
                }
                mv[j] = v; mi[j] = id;
            }
        }
        #pragma unroll
        for (int j = 0; j < 10; ++j) midx[g * 10 + j] = mi[j];
    }
}

// ---------------------------------------------------------------------------
// K5: resolve multi-match + fused GIoU / obj-BCE / cls-BCE loss.
// Deterministic block partials (no atomics): part[(b*LOSS_BLOCKS+blk)*4+j].
// ---------------------------------------------------------------------------
__global__ void __launch_bounds__(256)
k_loss(const float* __restrict__ p8, const float* __restrict__ p16,
       const float* __restrict__ p32, const float* __restrict__ labels, int b,
       const float4* __restrict__ bbox, const float* __restrict__ iou,
       const float* __restrict__ cost, const unsigned* __restrict__ fgc,
       const int* __restrict__ dynk, const int* __restrict__ midx,
       float* __restrict__ part)
{
    int a = blockIdx.x * blockDim.x + threadIdx.x;
    float liou = 0.f, lobj = 0.f, lcls = 0.f, nf = 0.f;
    if (a < NUM_A) {
        unsigned mask = 0;
        if (fgc[a]) {
            for (int g = 0; g < NUM_G; ++g) {
                int dk = dynk[g];
                const int* mi = midx + g * 10;
                #pragma unroll
                for (int j = 0; j < 10; ++j)
                    if (j < dk && mi[j] == a) mask |= 1u << g;
            }
        }
        if (__popc(mask) > 1) {
            float bc = INFINITY; int bg = 0;
            for (int g = 0; g < NUM_G; ++g) {
                float c = cost[(size_t)g * NUM_A + a];
                if (c < bc) { bc = c; bg = g; }   // first-min ties like argmin
            }
            mask = 1u << bg;
        }
        bool fg = mask != 0;
        const float* base; int i, HW;
        if (a < A0)      { base = p8  + (size_t)b * 85 * 6400; i = a;      HW = 6400; }
        else if (a < A1) { base = p16 + (size_t)b * 85 * 1600; i = a - A0; HW = 1600; }
        else             { base = p32 + (size_t)b * 85 * 400;  i = a - A1; HW = 400;  }

        float fgf = fg ? 1.f : 0.f;
        lobj = bce(base[4 * HW + i], fgf);
        if (fg) {
            int mg = __ffs(mask) - 1;
            float piou = iou[(size_t)mg * NUM_A + a];
            const float* L = labels + ((size_t)b * NUM_G + mg) * 5;
            float gx = L[0], gy = L[1], gw = L[2], gh = L[3];
            int mcls = (int)L[4];
            float4 bb = bbox[a];
            float tlx = fmaxf(bb.x - 0.5f * bb.z, gx - 0.5f * gw);
            float tly = fmaxf(bb.y - 0.5f * bb.w, gy - 0.5f * gh);
            float brx = fminf(bb.x + 0.5f * bb.z, gx + 0.5f * gw);
            float bry = fminf(bb.y + 0.5f * bb.w, gy + 0.5f * gh);
            float ab = bb.z * bb.w, ag = gw * gh;
            float inter = (tlx < brx && tly < bry) ? (brx - tlx) * (bry - tly) : 0.f;
            float uni  = ab + ag - inter;
            float iv   = inter / (uni + 1e-16f);
            float ctlx = fminf(bb.x - 0.5f * bb.z, gx - 0.5f * gw);
            float ctly = fminf(bb.y - 0.5f * bb.w, gy - 0.5f * gh);
            float cbrx = fmaxf(bb.x + 0.5f * bb.z, gx + 0.5f * gw);
            float cbry = fmaxf(bb.y + 0.5f * bb.w, gy + 0.5f * gh);
            float ac   = fmaxf((cbrx - ctlx) * (cbry - ctly), 1e-16f);
            float gi   = iv - (ac - uni) / ac;
            liou = 1.f - fminf(fmaxf(gi, -1.f), 1.f);
            for (int c = 0; c < NUM_C; ++c) {
                float x = base[(5 + c) * HW + i];
                float t = (c == mcls) ? piou : 0.f;
                lcls += bce(x, t);
            }
            nf = 1.f;
        }
    }
    __shared__ float s[256][4];
    int t = threadIdx.x;
    s[t][0] = liou; s[t][1] = lobj; s[t][2] = lcls; s[t][3] = nf;
    __syncthreads();
    for (int stp = 128; stp > 0; stp >>= 1) {
        if (t < stp) {
            s[t][0] += s[t + stp][0]; s[t][1] += s[t + stp][1];
            s[t][2] += s[t + stp][2]; s[t][3] += s[t + stp][3];
        }
        __syncthreads();
    }
    if (t == 0) {
        float* pp = part + ((size_t)b * LOSS_BLOCKS + blockIdx.x) * 4;
        pp[0] = s[0][0]; pp[1] = s[0][1]; pp[2] = s[0][2]; pp[3] = s[0][3];
    }
}

// ---------------------------------------------------------------------------
// K6: deterministic final reduction of block partials.
// ---------------------------------------------------------------------------
__global__ void k_final(const float* __restrict__ part, float* __restrict__ out)
{
    if (threadIdx.x == 0 && blockIdx.x == 0) {
        float li = 0.f, lo = 0.f, lc = 0.f, nf = 0.f;
        for (int k = 0; k < NUM_B * LOSS_BLOCKS; ++k) {
            li += part[k * 4 + 0]; lo += part[k * 4 + 1];
            lc += part[k * 4 + 2]; nf += part[k * 4 + 3];
        }
        out[0] = (5.f * li + lo + lc) / fmaxf(nf, 1.f);
    }
}

extern "C" void kernel_launch(void* const* d_in, const int* in_sizes, int n_in,
                              void* d_out, int out_size, void* d_ws, size_t ws_size,
                              hipStream_t stream)
{
    (void)in_sizes; (void)n_in; (void)out_size; (void)ws_size;
    const float* p8     = (const float*)d_in[0];
    const float* p16    = (const float*)d_in[1];
    const float* p32    = (const float*)d_in[2];
    const float* labels = (const float*)d_in[3];

    char* w = (char*)d_ws;
    size_t off = 0;
    auto alloc = [&](size_t bytes, size_t align) -> char* {
        off = (off + align - 1) & ~(align - 1);
        char* p = w + off; off += bytes; return p;
    };
    float4*   bbox = (float4*)  alloc((size_t)NUM_A * sizeof(float4), 16);
    float*    slp  = (float*)   alloc((size_t)NUM_A * sizeof(float), 16);
    _Float16* Dh   = (_Float16*)alloc((size_t)NUM_A * C_PAD * sizeof(_Float16), 32);
    _Float16* Th   = (_Float16*)alloc((size_t)32 * C_PAD * sizeof(_Float16), 32);
    float*    iou  = (float*)   alloc((size_t)NUM_G * NUM_A * sizeof(float), 16);
    float*    cost = (float*)   alloc((size_t)NUM_G * NUM_A * sizeof(float), 16);
    unsigned* inb  = (unsigned*)alloc((size_t)NUM_A * sizeof(unsigned), 16);
    unsigned* fgc  = (unsigned*)alloc((size_t)NUM_A * sizeof(unsigned), 16);
    int*      dynk = (int*)     alloc((size_t)NUM_G * sizeof(int), 16);
    int*      midx = (int*)     alloc((size_t)NUM_G * 10 * sizeof(int), 16);
    float*    part = (float*)   alloc((size_t)NUM_B * LOSS_BLOCKS * 4 * sizeof(float), 16);

    for (int b = 0; b < NUM_B; ++b) {
        k_decode<<<(NUM_A + 255) / 256, 256, 0, stream>>>(
            p8, p16, p32, labels, b, bbox, slp, Dh, iou, inb, fgc);
        k_onehot<<<1, 32, 0, stream>>>(labels, b, Th);
        k_cost_wmma<<<NUM_A / 16, 32, 0, stream>>>(
            Th, Dh, slp, iou, inb, fgc, cost);
        k_dynk<<<NUM_G, 32, 0, stream>>>(iou, fgc, dynk);
        k_sel<<<NUM_G, 32, 0, stream>>>(cost, midx);
        k_loss<<<(NUM_A + 255) / 256, 256, 0, stream>>>(
            p8, p16, p32, labels, b, bbox, iou, cost, fgc, dynk, midx, part);
    }
    k_final<<<1, 1, 0, stream>>>(part, (float*)d_out);
}
